// TokenModel_1030792151725
// MI455X (gfx1250) — matrix operand order repaired and checked by reading, not compiled, not run on hardware
//
#include <hip/hip_runtime.h>
#include <hip/hip_bf16.h>
#include <math.h>

typedef __attribute__((ext_vector_type(16))) _Float16 v16h;
typedef __attribute__((ext_vector_type(8)))  float    v8f;

#define Bn 64
#define Sn 512
#define Hn 768
#define N1 100   // token head hidden width
#define NP 128   // padded width (8 waves x 16)
#define N2 300   // sentence head hidden width

// ---------------------------------------------------------------------------
// K0: pack w1 [768,100] f32 into w1t [128][768] f16, transposed + zero-padded,
// so each lane's B fragment (col n, 16 consecutive K) is one contiguous 32B load.
// ---------------------------------------------------------------------------
__global__ __launch_bounds__(256) void prep_w1_kernel(
    const float* __restrict__ w1, _Float16* __restrict__ w1t)
{
    const int idx  = blockIdx.x * 256 + threadIdx.x;   // 0 .. 128*768-1
    const int ncol = idx / Hn;
    const int k    = idx % Hn;
    const float v  = (ncol < N1) ? w1[(size_t)k * N1 + ncol] : 0.0f;
    w1t[idx] = (_Float16)v;
}

// ---------------------------------------------------------------------------
// K1: tok = sigmoid( tanh(X @ w1 + b1) @ w2 + b2 )   X = hidden as [32768,768]
// One block = 16 rows. 8 waves, wave w owns N-columns [16w, 16w+16).
// K-loop: v_wmma_f32_16x16x32_f16; A converted f32->f16 from hidden, B from
// pre-packed w1t with straight-line vector loads (no exec-mask branches).
// ---------------------------------------------------------------------------
__global__ __launch_bounds__(256) void tok_head_kernel(
    const float* __restrict__ X,
    const _Float16* __restrict__ w1t, const float* __restrict__ b1,
    const float* __restrict__ w2, const float* __restrict__ b2,
    float* __restrict__ tok)
{
    const int tid  = threadIdx.x;
    const int lane = tid & 31;
    const int wave = tid >> 5;
    const int lrow = lane & 15;
    const int lhi  = lane >> 4;          // half-wave select
    const int m0   = blockIdx.x * 16;    // row tile base
    const int n0   = wave * 16;          // col tile base (0..112)
    const int m    = m0 + lrow;          // A row for this lane
    const int n    = n0 + lrow;          // B/C column for this lane (< 128)

    const float*    arow = X + (size_t)m * Hn;
    const _Float16* bcol = w1t + (size_t)n * Hn;

    v8f c = {};
    for (int kk = 0; kk < Hn; kk += 32) {
        // ---- A fragment: 16x32 f16 (contiguous k order, same as B)
        // lanes 0-15:  K = kk + {0..15};  lanes 16-31: kk + {16..31}
        v16h a;
        {
            const float* p = arow + kk + lhi * 16;
            float4 q0 = *reinterpret_cast<const float4*>(p);
            float4 q1 = *reinterpret_cast<const float4*>(p + 4);
            float4 q2 = *reinterpret_cast<const float4*>(p + 8);
            float4 q3 = *reinterpret_cast<const float4*>(p + 12);
            a[0]  = (_Float16)q0.x; a[1]  = (_Float16)q0.y;
            a[2]  = (_Float16)q0.z; a[3]  = (_Float16)q0.w;
            a[4]  = (_Float16)q1.x; a[5]  = (_Float16)q1.y;
            a[6]  = (_Float16)q1.z; a[7]  = (_Float16)q1.w;
            a[8]  = (_Float16)q2.x; a[9]  = (_Float16)q2.y;
            a[10] = (_Float16)q2.z; a[11] = (_Float16)q2.w;
            a[12] = (_Float16)q3.x; a[13] = (_Float16)q3.y;
            a[14] = (_Float16)q3.z; a[15] = (_Float16)q3.w;
        }
        // ---- B fragment: 32x16 f16; lanes 0-15 hold K = kk..kk+15 of col n,
        // lanes 16-31 hold K = kk+16..kk+31. One contiguous 32B vector load.
        const v16h bf = *reinterpret_cast<const v16h*>(bcol + kk + lhi * 16);

        c = __builtin_amdgcn_wmma_f32_16x16x32_f16(
                /*neg_a=*/false, a, /*neg_b=*/false, bf,
                /*c_mod=*/(short)0, c, /*reuse_a=*/false, /*reuse_b=*/false);
    }

    // C layout: VGPR r -> (local row = r + 8*lhi, col = n). Fold bias, tanh,
    // and the 100->1 projection into per-row LDS accumulators.
    __shared__ float srow[16];
    if (tid < 16) srow[tid] = 0.0f;
    __syncthreads();

    if (n < N1) {
        const float wv = w2[n];
        const float bb = b1[n];
        #pragma unroll
        for (int r = 0; r < 8; ++r) {
            const int lm = r + 8 * lhi;
            float h = tanhf(c[r] + bb);
            atomicAdd(&srow[lm], h * wv);
        }
    }
    __syncthreads();

    if (tid < 16) {
        float x = srow[tid] + b2[0];
        tok[m0 + tid] = 1.0f / (1.0f + expf(-x));
    }
}

// ---------------------------------------------------------------------------
// K2: segmented max per word, mask to starts, row reductions.
// masked[b,s] = is_start ? max(tok over segment) : 0
// Also: norm = masked / rowsum, rowmax, min(ones_mask), max(label), token-loss
// ---------------------------------------------------------------------------
__global__ __launch_bounds__(512) void seg_kernel(
    const float* __restrict__ tok, const int* __restrict__ offmap,
    const int* __restrict__ labels,
    float* __restrict__ masked_out, float* __restrict__ norm,
    float* __restrict__ rowmax, float* __restrict__ rowmin,
    float* __restrict__ slab, float* __restrict__ tloss)
{
    const int b = blockIdx.x;
    const int s = threadIdx.x;
    const int*   om = offmap + (size_t)b * Sn * 2;
    const float* tr = tok    + (size_t)b * Sn;

    const bool start = (om[2 * s] == 0);
    float m_val = 0.0f;
    if (start) {
        float mx = tr[s];
        for (int j = s + 1; j < Sn && om[2 * j] != 0; ++j)
            mx = fmaxf(mx, tr[j]);
        m_val = mx;
    }
    const float lab  = (float)labels[(size_t)b * Sn + s];
    const float ones = (m_val == 0.0f) ? 1.0f : m_val;
    const float dsq  = (m_val - lab) * (m_val - lab);

    __shared__ float s_sum[Sn], s_max[Sn], s_min[Sn], s_lab[Sn], s_sq[Sn];
    s_sum[s] = m_val; s_max[s] = m_val; s_min[s] = ones;
    s_lab[s] = lab;   s_sq[s]  = dsq;
    __syncthreads();
    for (int off = Sn / 2; off > 0; off >>= 1) {
        if (s < off) {
            s_sum[s] += s_sum[s + off];
            s_max[s]  = fmaxf(s_max[s], s_max[s + off]);
            s_min[s]  = fminf(s_min[s], s_min[s + off]);
            s_lab[s]  = fmaxf(s_lab[s], s_lab[s + off]);
            s_sq[s]  += s_sq[s + off];
        }
        __syncthreads();
    }
    const float total = s_sum[0];
    masked_out[(size_t)b * Sn + s] = m_val;
    norm[(size_t)b * Sn + s]       = m_val / total;
    if (s == 0) {
        rowmax[b] = s_max[0];
        rowmin[b] = s_min[0];
        slab[b]   = s_lab[0];
        tloss[b]  = s_sq[0];
    }
}

// ---------------------------------------------------------------------------
// K3: pooled[b,h] = sum_s hidden[b,s,h] * norm[b,s]  (GEMV, bandwidth-bound)
// ---------------------------------------------------------------------------
__global__ __launch_bounds__(256) void pool_kernel(
    const float* __restrict__ X, const float* __restrict__ norm,
    float* __restrict__ pooled)
{
    const int b = blockIdx.y;
    const int h = blockIdx.x * 256 + threadIdx.x;

    __shared__ float ns[Sn];
    for (int s = threadIdx.x; s < Sn; s += 256)
        ns[s] = norm[(size_t)b * Sn + s];
    __syncthreads();

    const float* xb = X + (size_t)b * Sn * Hn + h;
    float acc = 0.0f;
    for (int s = 0; s < Sn; ++s)
        acc += xb[(size_t)s * Hn] * ns[s];
    pooled[(size_t)b * Hn + h] = acc;
}

// ---------------------------------------------------------------------------
// K4: sent[b] = sigmoid( tanh(pooled[b] @ sw1 + sb1) @ sw2 + sb2 )
// ---------------------------------------------------------------------------
__global__ __launch_bounds__(256) void sent_kernel(
    const float* __restrict__ pooled,
    const float* __restrict__ sw1, const float* __restrict__ sb1,
    const float* __restrict__ sw2, const float* __restrict__ sb2,
    float* __restrict__ sent)
{
    const int b = blockIdx.x, tid = threadIdx.x;
    __shared__ float ps[Hn];
    __shared__ float red[256];
    for (int k = tid; k < Hn; k += 256)
        ps[k] = pooled[(size_t)b * Hn + k];
    __syncthreads();

    float part = 0.0f;
    for (int n = tid; n < N2; n += 256) {
        float d = sb1[n];
        for (int k = 0; k < Hn; ++k)
            d += ps[k] * sw1[(size_t)k * N2 + n];
        part += tanhf(d) * sw2[n];
    }
    red[tid] = part;
    __syncthreads();
    for (int off = 128; off > 0; off >>= 1) {
        if (tid < off) red[tid] += red[tid + off];
        __syncthreads();
    }
    if (tid == 0) {
        float x = red[0] + sb2[0];
        sent[b] = 1.0f / (1.0f + expf(-x));
    }
}

// ---------------------------------------------------------------------------
// K5: final scalar losses.
// ---------------------------------------------------------------------------
__global__ __launch_bounds__(64) void loss_kernel(
    const float* __restrict__ sent, const float* __restrict__ rowmax,
    const float* __restrict__ rowmin, const float* __restrict__ slab,
    const float* __restrict__ tloss, float* __restrict__ out)
{
    const int b = threadIdx.x;   // 0..63
    float sl = sent[b] - slab[b];  sl *= sl;
    float tl = tloss[b];
    float ra = rowmin[b] * rowmin[b];
    float rb = rowmax[b] - slab[b]; rb *= rb;

    __shared__ float a0[64], a1[64], a2[64], a3[64];
    a0[b] = sl; a1[b] = tl; a2[b] = ra; a3[b] = rb;
    __syncthreads();
    for (int off = 32; off > 0; off >>= 1) {
        if (b < off) {
            a0[b] += a0[b + off]; a1[b] += a1[b + off];
            a2[b] += a2[b + off]; a3[b] += a3[b + off];
        }
        __syncthreads();
    }
    if (b == 0) {
        const float sentence_loss = a0[0], token_loss = a1[0];
        const float reg_a = a2[0], reg_b = a3[0];
        out[1] = sentence_loss;
        out[2] = token_loss;
        out[3] = reg_a;
        out[4] = reg_b;
        out[0] = 1.0f * sentence_loss + 1.0f * token_loss
               + 0.01f * (reg_a + reg_b);
    }
}

// ---------------------------------------------------------------------------
extern "C" void kernel_launch(void* const* d_in, const int* in_sizes, int n_in,
                              void* d_out, int out_size, void* d_ws, size_t ws_size,
                              hipStream_t stream) {
    const float* hidden = (const float*)d_in[0];
    const int*   offmap = (const int*)  d_in[1];
    const int*   labels = (const int*)  d_in[2];
    const float* w1  = (const float*)d_in[3];
    const float* b1  = (const float*)d_in[4];
    const float* w2  = (const float*)d_in[5];
    const float* b2  = (const float*)d_in[6];
    const float* sw1 = (const float*)d_in[7];
    const float* sb1 = (const float*)d_in[8];
    const float* sw2 = (const float*)d_in[9];
    const float* sb2 = (const float*)d_in[10];

    float* out = (float*)d_out;
    float* ws  = (float*)d_ws;

    // workspace layout (floats)
    float* tok    = ws;                    // 32768
    float* norm   = ws + 32768;            // 32768
    float* rowmax = ws + 65536;            // 64
    float* rowmin = ws + 65600;            // 64
    float* slab   = ws + 65664;            // 64
    float* tl     = ws + 65728;            // 64
    // w1t (128*768 f16 = 49152 floats) lives before K3; pooled reuses the
    // same region afterwards (K1 is done with w1t by the time K3 writes).
    _Float16* w1t = (_Float16*)(ws + 65792);
    float* pooled = ws + 65792;            // 49152 (reuse of w1t region)

    // output layout: [loss, sent_loss, tok_loss, reg_a, reg_b,
    //                 masked(64*512), sent(64)]
    float* masked_out = out + 5;
    float* sent       = out + 5 + Bn * Sn;

    prep_w1_kernel<<<(NP * Hn) / 256, 256, 0, stream>>>(w1, w1t);
    tok_head_kernel<<<(Bn * Sn) / 16, 256, 0, stream>>>(hidden, w1t, b1, w2, b2, tok);
    seg_kernel<<<Bn, Sn, 0, stream>>>(tok, offmap, labels, masked_out, norm,
                                      rowmax, rowmin, slab, tl);
    pool_kernel<<<dim3(Hn / 256, Bn), 256, 0, stream>>>(hidden, norm, pooled);
    sent_kernel<<<Bn, 256, 0, stream>>>(pooled, sw1, sb1, sw2, sb2, sent);
    loss_kernel<<<1, 64, 0, stream>>>(sent, rowmax, rowmin, slab, tl, out);
}
